// BertSelfOutput_59983513256155
// MI455X (gfx1250) — 8-bit matrix operand order repaired and checked by reading, compiled, not run on hardware
//
#include <hip/hip_runtime.h>

typedef __attribute__((ext_vector_type(8))) int v8i;

#define HDIM 1024
#define TILE_M 32
#define KCHUNKS 16
#define FRAG_DW 256

__device__ __forceinline__ float bf16r(float x) {
  unsigned u = __float_as_uint(x);
  u += 0x7FFFu + ((u >> 16) & 1u);
  return __uint_as_float(u & 0xFFFF0000u);
}

__device__ __forceinline__ int q8(float x, float inv) {
  int q = (int)rintf(x * inv);
  q = max(-128, min(127, q));
  return q & 0xFF;
}

__global__ void quant_weights_kernel(const float* __restrict__ W,
                                     const float* __restrict__ wscale,
                                     int* __restrict__ qW) {
  const int t = blockIdx.x * blockDim.x + threadIdx.x;
  const int frag = t >> 8;
  const int lane = (t >> 3) & 31;
  const int d    = t & 7;
  const int o_tile = frag >> 4;
  const int kc     = frag & 15;
  const int o    = (o_tile << 4) + (lane & 15);
  const int koff = ((d >> 2) << 5) + ((lane >> 4) << 4) + ((d & 3) << 2);
  const int k    = (kc << 6) + koff;
  const float inv = 1.0f / wscale[o];
  const float4 w = *(const float4*)(W + o * HDIM + k);
  qW[t] = q8(w.x, inv) | (q8(w.y, inv) << 8) | (q8(w.z, inv) << 16) |
          (q8(w.w, inv) << 24);
}

__global__ void __launch_bounds__(256)
fused_kernel(const float* __restrict__ X,
             const float* __restrict__ R,
             const int*   __restrict__ qW,
             const float* __restrict__ bias,
             const float* __restrict__ gamma,
             const float* __restrict__ beta,
             const float* __restrict__ act_scale,
             const float* __restrict__ wscale,
             float* __restrict__ out) {
  __shared__ int aQ[2 * KCHUNKS * FRAG_DW];
  __shared__ unsigned short hBuf[TILE_M * HDIM];

  const int m0  = blockIdx.x * TILE_M;
  const int tid = threadIdx.x;
  const float sA   = act_scale[0];
  const float invA = 1.0f / sA;

#pragma unroll
  for (int i = 0; i < 32; ++i) {
    const int idx = tid + (i << 8);
    const int m  = idx >> 8;
    const int k4 = (idx & 255) << 2;
    const float4 v = *(const float4*)(X + (m0 + m) * HDIM + k4);
    const int q = q8(v.x, invA) | (q8(v.y, invA) << 8) |
                  (q8(v.z, invA) << 16) | (q8(v.w, invA) << 24);
    const int kc   = k4 >> 6;
    const int koff = k4 & 63;
    const int half = (koff >> 4) & 1;
    const int d    = ((koff >> 5) << 2) + ((koff >> 2) & 3);
    const int lane = (half << 4) + (m & 15);
    aQ[((m >> 4) << 12) + (kc << 8) + (lane << 3) + d] = q;
  }
  __syncthreads();

  const int wave = tid >> 5;
  const int lane = tid & 31;
  const int nloc = lane & 15;
  const int mh   = (lane >> 4) << 3;

#pragma unroll 1
  for (int t = 0; t < 8; ++t) {
    const int o_tile = wave + (t << 3);
    const int* bbase = qW + ((o_tile << 4) << 8) + (lane << 3);
    const int* abase = aQ + (lane << 3);
    v8i acc0 = {};
    v8i acc1 = {};
#pragma unroll 4
    for (int kc = 0; kc < KCHUNKS; ++kc) {
      const v8i bfrag = *(const v8i*)(bbase + (kc << 8));
      const v8i a0 = *(const v8i*)(abase + (kc << 8));
      const v8i a1 = *(const v8i*)(abase + 4096 + (kc << 8));
      acc0 = __builtin_amdgcn_wmma_i32_16x16x64_iu8(true, a0, true, bfrag, acc0,
                                                    false, false);
      acc1 = __builtin_amdgcn_wmma_i32_16x16x64_iu8(true, a1, true, bfrag, acc1,
                                                    false, false);
    }
    const int n = (o_tile << 4) + nloc;
    const float sc = sA * wscale[n];
    const float bv = bias[n];
#pragma unroll
    for (int j = 0; j < 8; ++j) {
      const int m = mh + j;
      const float y0 = sc * (float)acc0[j] + bv;
      const float y1 = sc * (float)acc1[j] + bv;
      hBuf[m * HDIM + n]        = (unsigned short)(__float_as_uint(bf16r(y0)) >> 16);
      hBuf[(16 + m) * HDIM + n] = (unsigned short)(__float_as_uint(bf16r(y1)) >> 16);
    }
  }
  __syncthreads();

#pragma unroll 1
  for (int rr = 0; rr < 4; ++rr) {
    const int r = (wave << 2) + rr;
    unsigned int* row = (unsigned int*)(hBuf + r * HDIM);
    const float2* R2 = (const float2*)(R + (m0 + r) * HDIM);
    float sum = 0.0f;
#pragma unroll
    for (int i = 0; i < 16; ++i) {
      const int c2 = lane + (i << 5);
      const unsigned p = row[c2];
      const float2 rv = R2[c2];
      const float h0 = bf16r(__uint_as_float(p << 16)        + bf16r(rv.x));
      const float h1 = bf16r(__uint_as_float(p & 0xFFFF0000u) + bf16r(rv.y));
      row[c2] = (__float_as_uint(h0) >> 16) | (__float_as_uint(h1) & 0xFFFF0000u);
      sum += h0 + h1;
    }
#pragma unroll
    for (int off = 16; off > 0; off >>= 1) sum += __shfl_xor(sum, off, 32);
    const float mean = sum * (1.0f / HDIM);

    float ss = 0.0f;
#pragma unroll
    for (int i = 0; i < 16; ++i) {
      const unsigned p = row[lane + (i << 5)];
      const float a = __uint_as_float(p << 16) - mean;
      const float b = __uint_as_float(p & 0xFFFF0000u) - mean;
      ss += a * a + b * b;
    }
#pragma unroll
    for (int off = 16; off > 0; off >>= 1) ss += __shfl_xor(ss, off, 32);
    const float rs = rsqrtf(ss * (1.0f / HDIM) + 1e-12f);

    float2* out2 = (float2*)(out + (m0 + r) * HDIM);
#pragma unroll
    for (int i = 0; i < 16; ++i) {
      const int c2 = lane + (i << 5);
      const unsigned p = row[c2];
      const float h0 = __uint_as_float(p << 16);
      const float h1 = __uint_as_float(p & 0xFFFF0000u);
      const float2 g  = ((const float2*)gamma)[c2];
      const float2 be = ((const float2*)beta)[c2];
      float2 o;
      o.x = (h0 - mean) * rs * g.x + be.x;
      o.y = (h1 - mean) * rs * g.y + be.y;
      out2[c2] = o;
    }
  }
}

extern "C" void kernel_launch(void* const* d_in, const int* in_sizes, int n_in,
                              void* d_out, int out_size, void* d_ws, size_t ws_size,
                              hipStream_t stream) {
  const float* X      = (const float*)d_in[0];
  const float* R      = (const float*)d_in[1];
  const float* W      = (const float*)d_in[2];
  const float* bias   = (const float*)d_in[3];
  const float* gamma  = (const float*)d_in[4];
  const float* beta   = (const float*)d_in[5];
  const float* ascale = (const float*)d_in[6];
  const float* wscale = (const float*)d_in[7];
  float* out = (float*)d_out;
  int* qW = (int*)d_ws;

  const int M = in_sizes[0] / HDIM;

  quant_weights_kernel<<<(HDIM * HDIM / 4) / 256, 256, 0, stream>>>(W, wscale, qW);
  fused_kernel<<<M / TILE_M, 256, 0, stream>>>(X, R, qW, bias, gamma, beta,
                                               ascale, wscale, out);
}
